// BahdanauAttentionLayer_53781580480696
// MI455X (gfx1250) — matrix operand order repaired and checked by reading, not compiled, not run on hardware
//
#include <hip/hip_runtime.h>
#include <hip/hip_bf16.h>

// ---------------------------------------------------------------------------
// Bahdanau attention, MI455X (gfx1250, wave32, WMMA + TDM async tensor load).
// B=32, T=4096, D=H=512.  Flattened M = B*T = 131072 rows.
// ---------------------------------------------------------------------------

#define Bv 32
#define Tv 4096
#define Dv 512
#define Hv 512
#define MROWS (Bv * Tv)         // 131072
#define MTILE 64                // rows per tile in fused kernel
#define NTILES (MROWS / MTILE)  // 2048
#define NBLOCKS 1024            // persistent blocks (grid-stride over tiles)
#define ROWP 520                // padded LDS row length (halves)

typedef __attribute__((ext_vector_type(8)))  float        v8f;
typedef __attribute__((ext_vector_type(4)))  float        f32x4;
typedef __attribute__((ext_vector_type(2)))  unsigned int u32x2;
typedef __attribute__((ext_vector_type(4)))  unsigned int u32x4;
typedef __attribute__((ext_vector_type(4)))  int          i32x4;
typedef __attribute__((ext_vector_type(8)))  int          i32x8;
typedef __attribute__((ext_vector_type(16))) __bf16       v16bf;

union FragBF {
    v16bf v;
    uint4 u[2];
};

// tanh via co-executing TRANS ops (v_exp_f32 + v_rcp_f32):
//   tanh(x) = 1 - 2/(exp(2x)+1);  exp->inf => +1, exp->0 => -1 (correct sat).
__device__ __forceinline__ float fast_tanh(float x) {
    const float e = __expf(2.0f * x);
    const float r = __builtin_amdgcn_rcpf(e + 1.0f);
    return fmaf(-2.0f, r, 1.0f);
}

// ---------------------------------------------------------------------------
// TDM: DMA one 64-row x 512-col f32 tile of enc into LDS (tracked: TENSORcnt).
// D# group0: count=1 | lds_addr | global_addr[56:0] | type=2.
// D# group1: wg_mask=0, data_size=2(4B), tensor 512x64, tile 512x64, stride 512.
// Groups 2/3 zero => 2-D tensor.  Issued by one wave only.
// This toolchain's builtin is the 6-arg form (extra zero i32x8 before cpol).
// ---------------------------------------------------------------------------
__device__ __forceinline__ void tdm_load_tile(const float* gptr, unsigned ldsOff) {
    const unsigned long long ga = (unsigned long long)(const void*)gptr;
    const u32x4 g0 = {
        1u,                                       // count=1, user descriptor
        ldsOff,                                   // lds_addr (bytes)
        (unsigned)ga,                             // global_addr[31:0]
        (unsigned)(ga >> 32) | 0x80000000u        // addr[56:32] | type=2 (<<30)
    };
    const i32x8 g1 = {
        0x00020000,   // [15:0] wg_mask=0 ; [17:16] data_size=2 -> 4 bytes
        0x02000000,   // [79:48] tensor_dim0=512 (low16 in dw1[31:16])
        0x00400000,   // [111:80] tensor_dim1=64 (low16 in dw2[31:16])
        0x02000000,   // [127:112] tile_dim0=512
        64,           // [143:128] tile_dim1=64 rows
        512,          // [207:160] tensor_dim0_stride=512 elements
        0, 0
    };
    const i32x4 z4 = {0, 0, 0, 0};
    const i32x8 z8 = {0, 0, 0, 0, 0, 0, 0, 0};
    __builtin_amdgcn_tensor_load_to_lds(g0, g1, z4, z4, z8, 0);
}

// ---------------------------------------------------------------------------
// Kernel 1: dens1[b][h] = hidden[b]@W[:,h] + bias[h]   (32x512, tiny)
// ---------------------------------------------------------------------------
__global__ __launch_bounds__(256) void dens1_kernel(
    const float* __restrict__ hidden, const float* __restrict__ W,
    const float* __restrict__ bias, float* __restrict__ d1) {
    __shared__ float hS[Dv];
    const int b = blockIdx.x, tid = threadIdx.x;
    for (int i = tid; i < Dv; i += 256) hS[i] = hidden[b * Dv + i];
    __syncthreads();
    for (int c = tid; c < Hv; c += 256) {
        float acc = bias[c];
        #pragma unroll 4
        for (int k = 0; k < Dv; ++k) acc = fmaf(hS[k], W[k * Hv + c], acc);
        d1[b * Hv + c] = acc;
    }
}

// ---------------------------------------------------------------------------
// Kernel 2: pack a 512x512 f32 weight into BF16 WMMA-B fragment-major layout.
// Fragment (kt, nt): 32 K-rows x 16 N-cols.  Lane n<16 holds col N=n, K=0..7,16..23;
// lane n+16 holds col N=n, K=8..15,24..31 (ISA 7.12.2 B layout).  Each lane's 16
// halves stored contiguously (32 B) -> fragment loads are two b128 per lane.
// ---------------------------------------------------------------------------
__global__ __launch_bounds__(32) void pack_bf16_kernel(
    const float* __restrict__ src, __bf16* __restrict__ dst) {
    const int frag = blockIdx.x;            // 16 kt * 32 nt = 512 frags
    const int lane = threadIdx.x;           // 32
    const int kt = frag >> 5, nt = frag & 31;
    const int l16 = lane & 15, hi = lane >> 4;
    const int col = nt * 16 + l16;
    __bf16 tmp[16];
    #pragma unroll
    for (int e = 0; e < 16; ++e)
        tmp[e] = (__bf16)src[(kt * 32 + hi * 8 + (e & 7) + ((e >> 3) << 4)) * Hv + col];
    uint4* d = (uint4*)(dst + ((long)frag * 32 + lane) * 16);
    d[0] = *(uint4*)&tmp[0];
    d[1] = *(uint4*)&tmp[8];
}

// ---------------------------------------------------------------------------
// Kernel 3 (core): persistent, TDM-double-buffered fused kernel:
//   pre = enc@U + dens1 ;  t = tanh(pre@Wa + ba) ;  cij = t @ V.
// Dynamic LDS layout (bytes):
//   stagF32 [64][512] f32 :      0 .. 131072   (TDM destination)
//   encS  [64][520] bf16  : 131072 .. 197632
//   preS  [64][520] bf16  : 197632 .. 264192
//   d1S   [512] f32       : 264192 .. 266240
//   vS    [512] f32       : 266240 .. 268288
//   baS   [512] f32       : 268288 .. 270336
//   cijS  [2][64] f32     : 270336 .. 270848
// ---------------------------------------------------------------------------
#define FUSED_LDS_BYTES 270848

__global__ __launch_bounds__(256) void attn_fused_kernel(
    const float*  __restrict__ enc,      // [B,T,D]
    const __bf16* __restrict__ Upack,    // packed fragments
    const __bf16* __restrict__ WaPack,   // packed fragments
    const float*  __restrict__ d1,       // [B,H] (bias folded in)
    const float*  __restrict__ ba,       // [H]
    const float*  __restrict__ Vvec,     // [H]
    float*        __restrict__ cij) {    // [B,T]
    extern __shared__ char smem[];
    float*  stag = (float*)smem;
    __bf16* encS = (__bf16*)(smem + MTILE * Dv * 4);
    __bf16* preS = encS + MTILE * ROWP;
    float*  d1S  = (float*)(preS + MTILE * ROWP);
    float*  vS   = d1S + Hv;
    float*  baS  = vS + Hv;
    float*  cijS = baS + Hv;

    const int tid  = threadIdx.x;
    const int w    = tid >> 5;                  // wave 0..7
    const int lane = tid & 31;
    const int l16  = lane & 15;
    const int hi   = lane >> 4;
    const int rt   = w & 3;                     // 16-row tile within 64
    const int cg   = w >> 2;                    // column group 0/1

    // LDS byte offset of the TDM staging buffer (flat LDS addr, low 32 bits).
    const unsigned stagOff = (unsigned)(size_t)(void*)stag;

    for (int i = tid; i < Hv; i += 256) {
        vS[i]  = Vvec[i];
        baS[i] = ba[i];
    }

    int tile = blockIdx.x;
    if (tile < NTILES && w == 0)
        tdm_load_tile(enc + (long)tile * MTILE * Dv, stagOff);

    for (; tile < NTILES; tile += NBLOCKS) {
        const long rowBase = (long)tile * MTILE;
        const int  b       = (int)(rowBase >> 12);   // tile never crosses batch

        for (int i = tid; i < Hv; i += 256) d1S[i] = d1[b * Hv + i];

        if (w == 0) __builtin_amdgcn_s_wait_tensorcnt(0);
        __syncthreads();                              // staged f32 tile visible

        // ---- convert staged f32 tile -> padded bf16 A-panel (LDS -> LDS) ----
        for (int i = tid * 4; i < MTILE * Dv; i += 256 * 4) {
            const int r = i >> 9, c = i & 511;
            const f32x4 f = *(const f32x4*)(stag + i);
            union { __bf16 h[4]; u32x2 u; } pk;
            pk.h[0] = (__bf16)f.x; pk.h[1] = (__bf16)f.y;
            pk.h[2] = (__bf16)f.z; pk.h[3] = (__bf16)f.w;
            *(u32x2*)(encS + r * ROWP + c) = pk.u;
        }
        __syncthreads();                              // staging consumed

        // ---- TDM-prefetch next tile while we do both GEMMs on this one ----
        const int nxt = tile + NBLOCKS;
        if (nxt < NTILES && w == 0)
            tdm_load_tile(enc + (long)nxt * MTILE * Dv, stagOff);

        // ---- GEMM1: preS = encS @ Upack + dens1 (bf16 WMMA, f32 accum) ----
        for (int cc = cg; cc < 8; cc += 2) {          // 64-col chunk
            v8f acc[4];
            #pragma unroll
            for (int n = 0; n < 4; ++n) {
                const float c0 = d1S[cc * 64 + n * 16 + l16];
                acc[n] = (v8f){c0, c0, c0, c0, c0, c0, c0, c0};
            }
            for (int kt = 0; kt < 16; ++kt) {         // K = 512, step 32
                FragBF a;
                const __bf16* ar = encS + (rt * 16 + l16) * ROWP + kt * 32 + hi * 8;
                a.u[0] = *(const uint4*)(ar);
                a.u[1] = *(const uint4*)(ar + 16);
                #pragma unroll
                for (int n = 0; n < 4; ++n) {
                    const int nt = cc * 4 + n;
                    FragBF bf;
                    const __bf16* bp = Upack + ((long)(kt * 32 + nt) * 32 + lane) * 16;
                    bf.u[0] = *(const uint4*)(bp);
                    bf.u[1] = *(const uint4*)(bp + 8);
                    acc[n] = __builtin_amdgcn_wmma_f32_16x16x32_bf16(
                        false, a.v, false, bf.v, (short)0, acc[n], false, false);
                }
            }
            #pragma unroll
            for (int n = 0; n < 4; ++n)
                #pragma unroll
                for (int j = 0; j < 8; ++j) {
                    const int r = rt * 16 + j + 8 * hi;   // C layout: M = j + 8*hi
                    const int c = cc * 64 + n * 16 + l16; // N = lane%16
                    preS[r * ROWP + c] = (__bf16)acc[n][j];
                }
        }
        __syncthreads();

        // ---- GEMM2 + tanh + dot(V): cij rows ----
        float cp[8] = {0.f, 0.f, 0.f, 0.f, 0.f, 0.f, 0.f, 0.f};
        for (int cc = cg; cc < 8; cc += 2) {
            v8f acc[4];
            #pragma unroll
            for (int n = 0; n < 4; ++n) {
                const float c0 = baS[cc * 64 + n * 16 + l16];
                acc[n] = (v8f){c0, c0, c0, c0, c0, c0, c0, c0};
            }
            for (int kt = 0; kt < 16; ++kt) {
                FragBF a;
                const __bf16* ar = preS + (rt * 16 + l16) * ROWP + kt * 32 + hi * 8;
                a.u[0] = *(const uint4*)(ar);
                a.u[1] = *(const uint4*)(ar + 16);
                #pragma unroll
                for (int n = 0; n < 4; ++n) {
                    const int nt = cc * 4 + n;
                    FragBF bf;
                    const __bf16* bp = WaPack + ((long)(kt * 32 + nt) * 32 + lane) * 16;
                    bf.u[0] = *(const uint4*)(bp);
                    bf.u[1] = *(const uint4*)(bp + 8);
                    acc[n] = __builtin_amdgcn_wmma_f32_16x16x32_bf16(
                        false, a.v, false, bf.v, (short)0, acc[n], false, false);
                }
            }
            #pragma unroll
            for (int n = 0; n < 4; ++n) {
                const float vv = vS[cc * 64 + n * 16 + l16];
                #pragma unroll
                for (int j = 0; j < 8; ++j)
                    cp[j] = fmaf(fast_tanh(acc[n][j]), vv, cp[j]);
            }
        }
        // Butterfly-reduce across the 16 lanes of each half (xor<16 keeps hi).
        #pragma unroll
        for (int m = 1; m < 16; m <<= 1)
            #pragma unroll
            for (int j = 0; j < 8; ++j)
                cp[j] += __shfl_xor(cp[j], m, 32);
        if (l16 == 0)
            #pragma unroll
            for (int j = 0; j < 8; ++j)
                cijS[cg * 64 + rt * 16 + hi * 8 + j] = cp[j];
        __syncthreads();
        if (tid < MTILE)
            cij[rowBase + tid] = cijS[tid] + cijS[64 + tid];
    }
}

// ---------------------------------------------------------------------------
// Kernel 4: row softmax over T per batch -> alphas
// ---------------------------------------------------------------------------
__global__ __launch_bounds__(256) void softmax_kernel(
    const float* __restrict__ cij, float* __restrict__ alphas) {
    __shared__ float red[256];
    const int b = blockIdx.x, tid = threadIdx.x;
    const float* row = cij + (long)b * Tv;
    float m = -INFINITY;
    for (int i = tid; i < Tv; i += 256) m = fmaxf(m, row[i]);
    red[tid] = m; __syncthreads();
    for (int s = 128; s > 0; s >>= 1) {
        if (tid < s) red[tid] = fmaxf(red[tid], red[tid + s]);
        __syncthreads();
    }
    m = red[0]; __syncthreads();
    float sum = 0.f;
    for (int i = tid; i < Tv; i += 256) sum += __expf(row[i] - m);
    red[tid] = sum; __syncthreads();
    for (int s = 128; s > 0; s >>= 1) {
        if (tid < s) red[tid] += red[tid + s];
        __syncthreads();
    }
    const float inv = 1.0f / red[0];
    for (int i = tid; i < Tv; i += 256)
        alphas[(long)b * Tv + i] = __expf(row[i] - m) * inv;
}

// ---------------------------------------------------------------------------
// Kernel 5: partial weighted sum  partial[b][tc][d] = sum_{t in chunk} a*enc
// enc is streamed exactly once -> non-temporal loads keep L2 for weights.
// ---------------------------------------------------------------------------
__global__ __launch_bounds__(256) void wsum_partial_kernel(
    const float* __restrict__ enc, const float* __restrict__ alphas,
    float* __restrict__ partial) {
    const int b = blockIdx.x, tc = blockIdx.y, tid = threadIdx.x;
    float a0 = 0.f, a1 = 0.f;
    for (int tt = 0; tt < Tv / 32; ++tt) {
        const int t = tc * (Tv / 32) + tt;
        const float al = alphas[(long)b * Tv + t];
        const float* er = enc + ((long)b * Tv + t) * Dv;
        a0 = fmaf(al, __builtin_nontemporal_load(er + tid), a0);
        a1 = fmaf(al, __builtin_nontemporal_load(er + tid + 256), a1);
    }
    float* p = partial + ((long)(b * 32 + tc)) * Dv;
    p[tid] = a0; p[tid + 256] = a1;
}

// ---------------------------------------------------------------------------
// Kernel 6: reduce the 32 T-chunks -> output[b][d]
// ---------------------------------------------------------------------------
__global__ __launch_bounds__(256) void wsum_reduce_kernel(
    const float* __restrict__ partial, float* __restrict__ out) {
    const int b = blockIdx.x, tid = threadIdx.x;
    float a0 = 0.f, a1 = 0.f;
    for (int tc = 0; tc < 32; ++tc) {
        const float* p = partial + ((long)(b * 32 + tc)) * Dv;
        a0 += p[tid]; a1 += p[tid + 256];
    }
    out[(long)b * Dv + tid] = a0;
    out[(long)b * Dv + tid + 256] = a1;
}

// ---------------------------------------------------------------------------
// Host launcher
// ---------------------------------------------------------------------------
extern "C" void kernel_launch(void* const* d_in, const int* in_sizes, int n_in,
                              void* d_out, int out_size, void* d_ws, size_t ws_size,
                              hipStream_t stream) {
    const float* hidden = (const float*)d_in[0];   // [B,D]
    const float* enc    = (const float*)d_in[1];   // [B,T,D]
    const float* W      = (const float*)d_in[2];   // [D,H]
    const float* U      = (const float*)d_in[3];   // [D,H]
    const float* Vvec   = (const float*)d_in[4];   // [H]
    const float* bias   = (const float*)d_in[5];   // [H]
    const float* Wa     = (const float*)d_in[6];   // [H,H]
    const float* ba     = (const float*)d_in[7];   // [H]

    float* out    = (float*)d_out;                 // [B,D]   (16384)
    float* alphas = (float*)d_out + Bv * Dv;       // [B,T]   (131072)

    // Workspace carve-up (all 16B aligned).
    char* ws = (char*)d_ws;
    float*  d1      = (float*) (ws + 0);                      //  64 KB
    __bf16* Upack   = (__bf16*)(ws + 65536);                  // 512 KB
    __bf16* WaPack  = (__bf16*)(ws + 65536 + 524288);         // 512 KB
    float*  cij     = (float*) (ws + 65536 + 2 * 524288);     // 512 KB
    float*  partial = (float*) (ws + 65536 + 3 * 524288);     //   2 MB

    dens1_kernel<<<Bv, 256, 0, stream>>>(hidden, W, bias, d1);
    pack_bf16_kernel<<<512, 32, 0, stream>>>(U, Upack);
    pack_bf16_kernel<<<512, 32, 0, stream>>>(Wa, WaPack);

    attn_fused_kernel<<<NBLOCKS, 256, FUSED_LDS_BYTES, stream>>>(
        enc, Upack, WaPack, d1, ba, Vvec, cij);

    softmax_kernel<<<Bv, 256, 0, stream>>>(cij, alphas);
    wsum_partial_kernel<<<dim3(Bv, 32), 256, 0, stream>>>(enc, alphas, partial);
    wsum_reduce_kernel<<<Bv, 256, 0, stream>>>(partial, out);
}
